// SingleHeadAttention_25786983645696
// MI455X (gfx1250) — matrix operand order repaired and checked by reading, not compiled, not run on hardware
//
#include <hip/hip_runtime.h>

#define SEQ   4096
#define BATCH 4
#define HID   768
#define HD    64

typedef __attribute__((ext_vector_type(16))) _Float16 v16h;
typedef __attribute__((ext_vector_type(8)))  _Float16 v8h;
typedef __attribute__((ext_vector_type(8)))  float    v8f;

union V16 {
  v16h v;
  struct { v8h lo, hi; } p;
};

__device__ __forceinline__ v8f wmma_f16(v16h a, v16h b, v8f c) {
  // D = A(16x32 f16) * B(32x16 f16) + C(16x16 f32)
  return __builtin_amdgcn_wmma_f32_16x16x32_f16(
      /*neg_a=*/false, a, /*neg_b=*/false, b,
      /*c_mod=*/(short)0, c, /*reuse_a=*/false, /*reuse_b=*/false);
}

// ---------------------------------------------------------------------------
// Kernel 1: transpose + convert W[768][64] f32 -> WT[64][768] f16 (x3)
// ---------------------------------------------------------------------------
__global__ void wprep_kernel(const float* __restrict__ Wq, const float* __restrict__ Wk,
                             const float* __restrict__ Wv,
                             _Float16* __restrict__ WTq, _Float16* __restrict__ WTk,
                             _Float16* __restrict__ WTv) {
  int i = blockIdx.x * blockDim.x + threadIdx.x;
  if (i >= HID * HD) return;
  int kk = i / HD, n = i % HD;
  WTq[n * HID + kk] = (_Float16)Wq[i];
  WTk[n * HID + kk] = (_Float16)Wk[i];
  WTv[n * HID + kk] = (_Float16)Wv[i];
}

// ---------------------------------------------------------------------------
// Kernel 2: projection, one output matrix per block (blockIdx.y = q/k/v).
// One wave computes a 16-row x 64-col tile. 4 accumulators + double-buffered
// A (x, f32->f16) and B (weight) fragments: ~130 VGPRs, no spills, 3072 waves.
// ---------------------------------------------------------------------------
__device__ __forceinline__ void loadA(const float* __restrict__ xrow, int kk, int h, V16& a) {
  v8f xlo = *(const v8f*)(xrow + kk + 16 * h);
  v8f xhi = *(const v8f*)(xrow + kk + 16 * h + 8);
#pragma unroll
  for (int i = 0; i < 8; ++i) {
    a.p.lo[i] = (_Float16)xlo[i];
    a.p.hi[i] = (_Float16)xhi[i];
  }
}

__global__ void __launch_bounds__(128)
proj_kernel(const float* __restrict__ x,
            const _Float16* __restrict__ WTq, const _Float16* __restrict__ WTk,
            const _Float16* __restrict__ WTv,
            const float* __restrict__ bq, const float* __restrict__ bk,
            const float* __restrict__ bv,
            _Float16* __restrict__ q, _Float16* __restrict__ k,
            _Float16* __restrict__ vT) {
  const int lane = threadIdx.x & 31;
  const int wv   = threadIdx.x >> 5;
  const int tile = blockIdx.x * 4 + wv;   // 1024 tiles of 16 rows
  const int rb   = tile * 16;             // flat row base (b*SEQ + s)
  const int h    = lane >> 4;
  const int n    = lane & 15;
  const int which = blockIdx.y;           // 0=q, 1=k, 2=v

  const _Float16* WT   = (which == 0) ? WTq : (which == 1) ? WTk : WTv;
  const float*    bias = (which == 0) ? bq : (which == 1) ? bk : bv;

  v8f c[4] = {};
  const float* xrow = x + (size_t)(rb + n) * HID;

  auto wf = [&](int kk, int nt) -> const v16h* {
    return (const v16h*)(WT + (size_t)(nt * 16 + n) * HID + kk + 16 * h);
  };

  V16 a0, a1;
  v16h bbA[4], bbB[4];
  loadA(xrow, 0, h, a0);
#pragma unroll
  for (int nt = 0; nt < 4; ++nt) bbA[nt] = *wf(0, nt);

  for (int kk = 0; kk < HID; kk += 64) {
    // phase 0: consume a0/bbA, prefetch a1/bbB (K = kk+32)
    loadA(xrow, kk + 32, h, a1);                 // kk+32 <= 736, always valid
    bbB[0] = *wf(kk + 32, 0);  c[0] = wmma_f16(a0.v, bbA[0], c[0]);
    bbB[1] = *wf(kk + 32, 1);  c[1] = wmma_f16(a0.v, bbA[1], c[1]);
    bbB[2] = *wf(kk + 32, 2);  c[2] = wmma_f16(a0.v, bbA[2], c[2]);
    bbB[3] = *wf(kk + 32, 3);  c[3] = wmma_f16(a0.v, bbA[3], c[3]);

    // phase 1: consume a1/bbB, prefetch a0/bbA (K = kk+64)
    if (kk + 64 < HID) {
      loadA(xrow, kk + 64, h, a0);
      bbA[0] = *wf(kk + 64, 0);  c[0] = wmma_f16(a1.v, bbB[0], c[0]);
      bbA[1] = *wf(kk + 64, 1);  c[1] = wmma_f16(a1.v, bbB[1], c[1]);
      bbA[2] = *wf(kk + 64, 2);  c[2] = wmma_f16(a1.v, bbB[2], c[2]);
      bbA[3] = *wf(kk + 64, 3);  c[3] = wmma_f16(a1.v, bbB[3], c[3]);
    } else {
      c[0] = wmma_f16(a1.v, bbB[0], c[0]);
      c[1] = wmma_f16(a1.v, bbB[1], c[1]);
      c[2] = wmma_f16(a1.v, bbB[2], c[2]);
      c[3] = wmma_f16(a1.v, bbB[3], c[3]);
    }
  }

  const int bIdx  = rb / SEQ;
  const int sBase = rb % SEQ;
  if (which != 2) {
    _Float16* dst = (which == 0) ? q : k;
#pragma unroll
    for (int nt = 0; nt < 4; ++nt) {
      const int col = nt * 16 + n;
      const float add = bias[col];
#pragma unroll
      for (int j = 0; j < 8; ++j)
        dst[(size_t)(rb + j + 8 * h) * HD + col] = (_Float16)(c[nt][j] + add);
    }
  } else {
#pragma unroll
    for (int nt = 0; nt < 4; ++nt) {
      const int col = nt * 16 + n;
      const float add = bias[col];
#pragma unroll
      for (int j = 0; j < 8; ++j)
        vT[((size_t)bIdx * HD + col) * SEQ + (sBase + j + 8 * h)] =
            (_Float16)(c[nt][j] + add);
    }
  }
}

// ---------------------------------------------------------------------------
// Kernel 3: causal flash attention, software-pipelined.
// One wave owns a 16-row q-tile and streams keys in 32-wide steps:
// K fragments double-buffered across steps; V fragments are transients loaded
// after the softmax so the LDS P-transpose round-trip hides their latency.
// ---------------------------------------------------------------------------
__global__ void __launch_bounds__(128)
flash_attn_kernel(const _Float16* __restrict__ q,
                  const _Float16* __restrict__ k,
                  const _Float16* __restrict__ vT,
                  float* __restrict__ out) {
  __shared__ __align__(32) _Float16 pbuf[4 * 16 * 32];
  const int lane = threadIdx.x & 31;
  const int wv   = threadIdx.x >> 5;
  const int tile = blockIdx.x * 4 + wv;
  const int bIdx = tile / (SEQ / 16);
  const int qb   = (tile % (SEQ / 16)) * 16;
  const int h    = lane >> 4;
  const int n    = lane & 15;
  _Float16* pb   = pbuf + wv * (16 * 32);

  const float scale = 0.0360843918243516f;  // 1/sqrt(768)  (reference scales by hidden_dim)

  // Q A-fragments (head dim split into two K=32 chunks), kept resident
  V16 aq[2];
  const _Float16* qrow = q + (size_t)(bIdx * SEQ + qb + n) * HD;
#pragma unroll
  for (int c = 0; c < 2; ++c) {
    aq[c].p.lo = *(const v8h*)(qrow + 32 * c + 16 * h);
    aq[c].p.hi = *(const v8h*)(qrow + 32 * c + 16 * h + 8);
  }

  v8f o[4] = {};
  float m[8], l[8];
#pragma unroll
  for (int j = 0; j < 8; ++j) { m[j] = -1e30f; l[j] = 0.0f; }

  const _Float16* kb0 = k + (size_t)bIdx * SEQ * HD;
  const _Float16* vb0 = vT + (size_t)bIdx * HD * SEQ;
  const int kend = qb + 16;

  v16h kf0[4], kf1[4];

  auto loadKfrags = [&](v16h* kf, int kv) {
    const _Float16* kr0 = kb0 + (size_t)(kv + n) * HD + 16 * h;
    kf[0] = *(const v16h*)(kr0);
    kf[1] = *(const v16h*)(kr0 + 32);
    kf[2] = *(const v16h*)(kr0 + 16 * HD);
    kf[3] = *(const v16h*)(kr0 + 16 * HD + 32);
  };

  auto step = [&](const v16h* kf, int kv) {
    // ---- scores: two 16x16 tiles (keys kv..kv+15, kv+16..kv+31) ----
    v8f s0 = {}, s1 = {};
    s0 = wmma_f16(aq[0].v, kf[0], s0);
    s0 = wmma_f16(aq[1].v, kf[1], s0);
    s1 = wmma_f16(aq[0].v, kf[2], s1);
    s1 = wmma_f16(aq[1].v, kf[3], s1);
#pragma unroll
    for (int j = 0; j < 8; ++j) { s0[j] *= scale; s1[j] *= scale; }

    if (kv + 31 > qb) {  // tail blocks crossing the diagonal
#pragma unroll
      for (int j = 0; j < 8; ++j) {
        const int row = qb + j + 8 * h;
        if (kv + n > row)      s0[j] = -1e30f;
        if (kv + 16 + n > row) s1[j] = -1e30f;
      }
    }

    // ---- online softmax (row stats; reduce across the 16 lanes of a half) ----
#pragma unroll
    for (int j = 0; j < 8; ++j) {
      float mm = fmaxf(s0[j], s1[j]);
      mm = fmaxf(mm, __shfl_xor(mm, 1, 32));
      mm = fmaxf(mm, __shfl_xor(mm, 2, 32));
      mm = fmaxf(mm, __shfl_xor(mm, 4, 32));
      mm = fmaxf(mm, __shfl_xor(mm, 8, 32));
      const float mn   = fmaxf(m[j], mm);
      const float corr = __expf(m[j] - mn);
      const float p0   = __expf(s0[j] - mn);
      const float p1   = __expf(s1[j] - mn);
      float rs = p0 + p1;
      rs += __shfl_xor(rs, 1, 32);
      rs += __shfl_xor(rs, 2, 32);
      rs += __shfl_xor(rs, 4, 32);
      rs += __shfl_xor(rs, 8, 32);
      l[j] = l[j] * corr + rs;
      m[j] = mn;
      const int row16 = j + 8 * h;
      pb[row16 * 32 + n]      = (_Float16)p0;
      pb[row16 * 32 + 16 + n] = (_Float16)p1;
#pragma unroll
      for (int nt = 0; nt < 4; ++nt) o[nt][j] *= corr;
    }

    // ---- V fragments: issued here so the LDS round-trip hides their latency
    const _Float16* vr = vb0 + (size_t)n * SEQ + kv + 16 * h;
    v16h vf0 = *(const v16h*)(vr);
    v16h vf1 = *(const v16h*)(vr + 16 * SEQ);
    v16h vf2 = *(const v16h*)(vr + 32 * SEQ);
    v16h vf3 = *(const v16h*)(vr + 48 * SEQ);

    // ---- P tile C-layout -> A-layout via LDS ----
    asm volatile("s_wait_dscnt 0" ::: "memory");
    V16 ap;
    ap.p.lo = *(const v8h*)(pb + n * 32 + 16 * h);
    ap.p.hi = *(const v8h*)(pb + n * 32 + 16 * h + 8);

    // ---- O += P * V ----
    o[0] = wmma_f16(ap.v, vf0, o[0]);
    o[1] = wmma_f16(ap.v, vf1, o[1]);
    o[2] = wmma_f16(ap.v, vf2, o[2]);
    o[3] = wmma_f16(ap.v, vf3, o[3]);
  };

  loadKfrags(kf0, 0);
  int kv = 0;
  while (true) {
    // even step: consumes kf0, prefetches kf1
    if (kv + 32 < kend) loadKfrags(kf1, kv + 32);
    if (kv + 256 + 16 <= SEQ) {
      __builtin_prefetch(kb0 + (size_t)(kv + 256 + n) * HD + 16 * h, 0, 3);
      __builtin_prefetch(vb0 + (size_t)n * SEQ + kv + 256 + 16 * h, 0, 3);
    }
    step(kf0, kv);
    kv += 32;
    if (kv >= kend) break;

    // odd step: consumes kf1, prefetches kf0
    if (kv + 32 < kend) loadKfrags(kf0, kv + 32);
    if (kv + 256 + 16 <= SEQ) {
      __builtin_prefetch(kb0 + (size_t)(kv + 256 + n) * HD + 16 * h, 0, 3);
      __builtin_prefetch(vb0 + (size_t)n * SEQ + kv + 256 + 16 * h, 0, 3);
    }
    step(kf1, kv);
    kv += 32;
    if (kv >= kend) break;
  }

  // ---- epilogue: normalize and store f32 output ----
#pragma unroll
  for (int j = 0; j < 8; ++j) {
    const float inv = 1.0f / l[j];
    const int row = bIdx * SEQ + qb + j + 8 * h;
#pragma unroll
    for (int nt = 0; nt < 4; ++nt)
      out[(size_t)row * HD + nt * 16 + n] = o[nt][j] * inv;
  }
}

// ---------------------------------------------------------------------------
extern "C" void kernel_launch(void* const* d_in, const int* in_sizes, int n_in,
                              void* d_out, int out_size, void* d_ws, size_t ws_size,
                              hipStream_t stream) {
  const float* x  = (const float*)d_in[0];
  const float* Wq = (const float*)d_in[1];
  const float* bq = (const float*)d_in[2];
  const float* Wk = (const float*)d_in[3];
  const float* bk = (const float*)d_in[4];
  const float* Wv = (const float*)d_in[5];
  const float* bv = (const float*)d_in[6];
  float* out = (float*)d_out;

  // Workspace carve-up (all f16, all 32B-aligned offsets): ~6.3 MB total
  _Float16* ws  = (_Float16*)d_ws;
  _Float16* WTq = ws;
  _Float16* WTk = WTq + (size_t)HID * HD;
  _Float16* WTv = WTk + (size_t)HID * HD;
  _Float16* qh  = WTv + (size_t)HID * HD;
  _Float16* kh  = qh + (size_t)BATCH * SEQ * HD;
  _Float16* vT  = kh + (size_t)BATCH * SEQ * HD;

  wprep_kernel<<<(HID * HD + 255) / 256, 256, 0, stream>>>(Wq, Wk, Wv, WTq, WTk, WTv);

  const int tiles = BATCH * SEQ / 16;  // 1024 row-tiles, 4 waves (128 thr) per block
  dim3 pgrid(tiles / 4, 3);            // y selects q / k / v -> 3072 waves
  proj_kernel<<<pgrid, 128, 0, stream>>>(x, WTq, WTk, WTv, bq, bk, bv, qh, kh, vT);
  flash_attn_kernel<<<tiles / 4, 128, 0, stream>>>(qh, kh, vT, out);
}
